// InvariantLayerj1_20564303413873
// MI455X (gfx1250) — matrix operand order repaired and checked by reading, not compiled, not run on hardware
//
#include <hip/hip_runtime.h>
#include <math.h>

typedef __attribute__((ext_vector_type(16))) __bf16 v16bf;
typedef __attribute__((ext_vector_type(8)))  float  v8f;

__device__ __forceinline__ unsigned short f2bf(float f) {
  unsigned u = __float_as_uint(f);
  unsigned r = (u + 0x7FFFu + ((u >> 16) & 1u)) >> 16;
  return (unsigned short)r;
}

__device__ __forceinline__ float magpair(float r, float i) {
  // inputs are unscaled quad combos; reference scales quads by 1/sqrt(2)
  return sqrtf(0.5f * (r * r + i * i));
}

// ---------------------------------------------------------------------------
// Kernel 0: pack weights A[f][k] (448x448 f32) into bf16 WMMA A-fragment order.
// Tile (ft,kt) is 16(f) x 32(k). Element (fr,kk):
//   g = kk>>3 ; lane = fr + 16*(g&1) ; e = (g>>1)*8 + (kk&7)
// flat: wp[((ft*14+kt)*32 + lane)*16 + e]
// ---------------------------------------------------------------------------
__global__ __launch_bounds__(256) void k_packW(const float* __restrict__ A,
                                               unsigned short* __restrict__ wp) {
  int t = blockIdx.x * 256 + threadIdx.x;
  if (t >= 28 * 14 * 512) return;
  int e    = t & 15;
  int lane = (t >> 4) & 31;
  int tile = t >> 9;
  int ft = tile / 14, kt = tile % 14;
  int laneHi = lane >> 4;
  int fr = lane & 15;
  int g  = ((e >> 3) << 1) | laneHi;
  int kk = (g << 3) | (e & 7);
  int f = ft * 16 + fr;
  int k = kt * 32 + kk;
  wp[t] = f2bf(A[f * 448 + k]);
}

// ---------------------------------------------------------------------------
// Kernel 1: DTCWT level-1 scattering for one (n, c, 16x16 output tile).
// Writes Z (bf16) directly in WMMA B-fragment order:
//   tile (mt,kt) is 32(k) x 16(m). Element (kk,mr):
//     lane = mr + 16*((kk>>3)&1) ; e = (kk>>4)*8 + (kk&7)
//   flat: zp[((mt*14+kt)*32 + lane)*16 + e]
//   m = n*16384 + h2*128 + w2  (so a 16-wide w2 run == one m-tile)
//   k = s*64 + c, s in {ll, lh1, hh1, hl1, hl2, hh2, lh2}
// ---------------------------------------------------------------------------
__global__ __launch_bounds__(256) void k_scat(const float* __restrict__ x,
                                              unsigned short* __restrict__ zp) {
  const int tileIdx = blockIdx.x;          // 0..63 : tw + 8*th
  const int c = blockIdx.y;                // 0..63
  const int n = blockIdx.z;                // 0..7
  const int tw = tileIdx & 7, th = tileIdx >> 3;
  const int W0 = tw * 16, H0 = th * 16;    // output-tile origin (out coords)
  const int tid = threadIdx.x;

  __shared__ float xs[38 * 40];            // input patch  (+/-3 halo, padded)
  __shared__ float lo[38 * 33];            // row-filtered (h0 along W)
  __shared__ float hi[38 * 33];            // row-filtered (h1 along W)

  const float h0c[5] = {-0.05f, 0.25f, 0.60f, 0.25f, -0.05f};
  const float h1c[7] = {-0.0107142857142857f, 0.0535714285714286f,
                         0.2607142857142857f, -0.6071428571428571f,
                         0.2607142857142857f, 0.0535714285714286f,
                        -0.0107142857142857f};

  const float* xc = x + ((size_t)(n * 64 + c)) * 65536;

  // Phase 1: load 38x38 mirrored patch (symmetric padding), NT loads
  for (int it = tid; it < 38 * 38; it += 256) {
    int ly = it / 38, lx = it - ly * 38;
    int gy = 2 * H0 - 3 + ly;
    int gx = 2 * W0 - 3 + lx;
    if (gy < 0) gy = -1 - gy; else if (gy >= 256) gy = 511 - gy;
    if (gx < 0) gx = -1 - gx; else if (gx >= 256) gx = 511 - gx;
    xs[ly * 40 + lx] = __builtin_nontemporal_load(xc + gy * 256 + gx);
  }
  __syncthreads();

  // Phase 2: row filter along W at 38 rows x 32 cols
  for (int it = tid; it < 38 * 32; it += 256) {
    int r = it >> 5, cx = it & 31;
    const float* row = &xs[r * 40 + cx];
    float aLo = 0.f, aHi = 0.f;
#pragma unroll
    for (int k = 0; k < 5; ++k) aLo += h0c[k] * row[k + 1];
#pragma unroll
    for (int k = 0; k < 7; ++k) aHi += h1c[k] * row[k];
    lo[r * 33 + cx] = aLo;
    hi[r * 33 + cx] = aHi;
  }
  __syncthreads();

  // Phase 3: column filter + q2c + pool, one output position per thread
  const int ty = tid >> 4, tx = tid & 15;
  float llv[2][2], lhv[2][2], hlv[2][2], hhv[2][2];
#pragma unroll
  for (int dy = 0; dy < 2; ++dy) {
#pragma unroll
    for (int dx = 0; dx < 2; ++dx) {
      int qy = 2 * ty + dy, qx = 2 * tx + dx;
      float sll = 0.f, slh = 0.f, shl = 0.f, shh = 0.f;
#pragma unroll
      for (int k = 0; k < 5; ++k) {
        float l5 = lo[(qy + 1 + k) * 33 + qx];
        float h5 = hi[(qy + 1 + k) * 33 + qx];
        sll += h0c[k] * l5;
        shl += h0c[k] * h5;
      }
#pragma unroll
      for (int k = 0; k < 7; ++k) {
        float l7 = lo[(qy + k) * 33 + qx];
        float h7 = hi[(qy + k) * 33 + qx];
        slh += h1c[k] * l7;
        shh += h1c[k] * h7;
      }
      llv[dy][dx] = sll; lhv[dy][dx] = slh;
      hlv[dy][dx] = shl; hhv[dy][dx] = shh;
    }
  }

  float z[7];
  z[0] = 0.25f * (llv[0][0] + llv[0][1] + llv[1][0] + llv[1][1]);
  // q2c: a=[0][0] b=[0][1] c=[1][0] d=[1][1]; band1=(a-d, b+c), band2=(a+d, b-c)
  z[1] = magpair(lhv[0][0] - lhv[1][1], lhv[0][1] + lhv[1][0]); // lh1
  z[2] = magpair(hhv[0][0] - hhv[1][1], hhv[0][1] + hhv[1][0]); // hh1
  z[3] = magpair(hlv[0][0] - hlv[1][1], hlv[0][1] + hlv[1][0]); // hl1
  z[4] = magpair(hlv[0][0] + hlv[1][1], hlv[0][1] - hlv[1][0]); // hl2
  z[5] = magpair(hhv[0][0] + hhv[1][1], hhv[0][1] - hhv[1][0]); // hh2
  z[6] = magpair(lhv[0][0] + lhv[1][1], lhv[0][1] - lhv[1][0]); // lh2

  const int h2 = H0 + ty;
  const int mt = n * 1024 + h2 * 8 + tw;   // m-tile index, mr = tx
#pragma unroll
  for (int s = 0; s < 7; ++s) {
    int kch = s * 64 + c;
    int kt = kch >> 5, kk = kch & 31;
    int lane = tx + (((kk >> 3) & 1) << 4);
    int e = ((kk >> 4) << 3) | (kk & 7);
    size_t addr = (((size_t)mt * 14 + kt) * 32 + lane) * 16 + e;
    zp[addr] = f2bf(z[s]);
  }
}

// ---------------------------------------------------------------------------
// Kernel 2: GEMM  out[n,f,h2,w2] = relu( sum_k W[f,k] * Z[k,m] + b[f] )
// Wave tile: 32(f) x 64(m) = 2 A-frags x 4 B-frags, K = 14 steps of 32.
// 8 x v_wmma_f32_16x16x32_bf16 per K step.
// ---------------------------------------------------------------------------
__global__ __launch_bounds__(256) void k_gemm(const unsigned short* __restrict__ wp_,
                                              const unsigned short* __restrict__ zp_,
                                              const float* __restrict__ bias,
                                              float* __restrict__ out) {
  const int wid  = threadIdx.x >> 5;
  const int lane = threadIdx.x & 31;
  const int gw = blockIdx.x * 8 + wid;     // 0..28671
  const int fb = gw % 14;                  // f-block of 32 (2 tiles)
  const int mb = gw / 14;                  // m-block of 64 (4 tiles)

  const __bf16* wp = (const __bf16*)wp_;
  const __bf16* zp = (const __bf16*)zp_;

  const v8f vzero = {0.f, 0.f, 0.f, 0.f, 0.f, 0.f, 0.f, 0.f};
  v8f acc[2][4];
#pragma unroll
  for (int i = 0; i < 2; ++i)
#pragma unroll
    for (int j = 0; j < 4; ++j) acc[i][j] = vzero;

  for (int kt = 0; kt < 14; ++kt) {
    v16bf af[2], bfr[4];
#pragma unroll
    for (int i = 0; i < 2; ++i) {
      int ft = fb * 2 + i;
      af[i] = *(const v16bf*)(wp + (((size_t)ft * 14 + kt) * 32 + lane) * 16);
    }
#pragma unroll
    for (int j = 0; j < 4; ++j) {
      int mt = mb * 4 + j;
      bfr[j] = *(const v16bf*)(zp + (((size_t)mt * 14 + kt) * 32 + lane) * 16);
    }
#pragma unroll
    for (int i = 0; i < 2; ++i)
#pragma unroll
      for (int j = 0; j < 4; ++j)
        acc[i][j] = __builtin_amdgcn_wmma_f32_16x16x32_bf16(
            false, af[i], false, bfr[j], (short)0, acc[i][j], false, false);
  }

  // D layout: VGPR r -> M(row f) = r (lanes 0-15) / r+8 (lanes 16-31); N = lane%16
  const int laneHi = lane >> 4;
  const int ncol   = lane & 15;
#pragma unroll
  for (int i = 0; i < 2; ++i) {
    int ft = fb * 2 + i;
#pragma unroll
    for (int j = 0; j < 4; ++j) {
      int mt = mb * 4 + j;
      int mbase = mt << 4;              // 16 consecutive m (same n, same h2 row)
      int nn  = mbase >> 14;
      int rem = mbase & 16383;          // h2*128 + w2_base
#pragma unroll
      for (int r = 0; r < 8; ++r) {
        int f = ft * 16 + r + 8 * laneHi;
        float v = acc[i][j][r] + bias[f];
        v = v > 0.f ? v : 0.f;
        size_t o = (((size_t)nn * 448 + f) << 14) + (size_t)(rem + ncol);
        __builtin_nontemporal_store(v, out + o);
      }
    }
  }
}

// ---------------------------------------------------------------------------
extern "C" void kernel_launch(void* const* d_in, const int* in_sizes, int n_in,
                              void* d_out, int out_size, void* d_ws, size_t ws_size,
                              hipStream_t stream) {
  const float* x = (const float*)d_in[0];   // (8,64,256,256) f32
  const float* A = (const float*)d_in[1];   // (448,448,1,1)  f32
  const float* b = (const float*)d_in[2];   // (448,)         f32
  float* out = (float*)d_out;               // (8,448,128,128) f32

  unsigned short* wp = (unsigned short*)d_ws;           // 200704 bf16 (401 KB)
  unsigned short* zpk = wp + 28 * 14 * 512;             // 58.7M bf16 (117.4 MB)

  (void)in_sizes; (void)n_in; (void)out_size; (void)ws_size;

  // pack weights -> bf16 A-fragments
  k_packW<<<(28 * 14 * 512 + 255) / 256, 256, 0, stream>>>(A, wp);

  // DTCWT scattering -> bf16 B-fragments
  dim3 g1(64, 64, 8);
  k_scat<<<g1, 256, 0, stream>>>(x, zpk);

  // WMMA GEMM + bias + relu
  k_gemm<<<3584, 256, 0, stream>>>(wp, zpk, b, out);
}
